// FlowSCConv_82240033783814
// MI455X (gfx1250) — matrix operand order repaired and checked by reading, not compiled, not run on hardware
//
#include <hip/hip_runtime.h>

#define N0C 200000
#define N1C 600000
#define N2C 400000
#define NGC 128

typedef __attribute__((ext_vector_type(16))) _Float16 v16h;
typedef __attribute__((ext_vector_type(8)))  float    v8f;

static inline int cdiv_l(long a, long b) { return (int)((a + b - 1) / b); }

// ---------------------------------------------------------------- zero
__global__ __launch_bounds__(256) void zero_f32(float* __restrict__ p, long n) {
  long i = (long)blockIdx.x * blockDim.x + threadIdx.x;
  if (i < n) p[i] = 0.0f;
}

// ---------------------------------------------------------------- dense proj: Y[N,32] = X[N,K] @ W[K,32]
// One wave per 16-row tile; two 16x16 WMMA accumulators cover the 32 output cols.
// N is guaranteed a multiple of 16 (N0/N1/N2 all are), so EXEC stays all-1s.
__global__ __launch_bounds__(256) void gemm_wmma_f16(
    const float* __restrict__ X, const float* __restrict__ W,
    float* __restrict__ Y, int N, int K)
{
  const int lane = threadIdx.x & 31;
  const int wave = threadIdx.x >> 5;
  const long base = ((long)blockIdx.x * 8 + wave) * 16;
  if (base >= N) return;                     // uniform per-wave exit
  const int half = lane >> 4;                // 0: lanes 0-15, 1: lanes 16-31
  const int m    = lane & 15;

  v8f c_lo = {0.f,0.f,0.f,0.f,0.f,0.f,0.f,0.f};
  v8f c_hi = {0.f,0.f,0.f,0.f,0.f,0.f,0.f,0.f};

  for (int kc = 0; kc < K; kc += 32) {
    // ---- A tile 16x32 f16: lane holds row m; half selects K-subranges
    //      elems 0..7  -> K = 16*half + 0..7
    //      elems 8..15 -> K = 16*half + 8 + 0..7   (contiguous, matches B)
    const float* xrow = X + (base + m) * (long)K + kc;
    const int kA0 = 16 * half;
    const int kA1 = 16 * half + 8;
    v16h a;
#pragma unroll
    for (int i = 0; i < 8; ++i) a[i]     = (_Float16)xrow[kA0 + i];
#pragma unroll
    for (int i = 0; i < 8; ++i) a[8 + i] = (_Float16)xrow[kA1 + i];

    // ---- B tiles 32x16 f16: lane holds column n; lanes 0-15 K=0..15, 16-31 K=16..31
    v16h b0, b1;
    const float* wk = W + (long)(kc + 16 * half) * 32;
#pragma unroll
    for (int i = 0; i < 16; ++i) {
      b0[i] = (_Float16)wk[i * 32 + m];
      b1[i] = (_Float16)wk[i * 32 + m + 16];
    }

    c_lo = __builtin_amdgcn_wmma_f32_16x16x32_f16(false, a, false, b0, (short)0, c_lo, false, false);
    c_hi = __builtin_amdgcn_wmma_f32_16x16x32_f16(false, a, false, b1, (short)0, c_hi, false, false);
  }

  // ---- D layout: VGPR j -> row = base + j + 8*half, col = m (c_lo) / m+16 (c_hi)
#pragma unroll
  for (int j = 0; j < 8; ++j) {
    long row = base + j + 8 * half;
    Y[row * 32 + m]      = c_lo[j];
    Y[row * 32 + m + 16] = c_hi[j];
  }
}

// ---------------------------------------------------------------- SpMM: acc[r,:] += v * Xp[c,:]
// One wave per nnz; lane = feature column -> coalesced 128B gather + 128B atomic scatter.
__global__ __launch_bounds__(256) void spmm_atomic(
    const int* __restrict__ rows, const int* __restrict__ cols,
    const float* __restrict__ vals, const float* __restrict__ Xp,
    float* __restrict__ acc, int nnz)
{
  const int lane = threadIdx.x & 31;
  long e = (long)blockIdx.x * 8 + (threadIdx.x >> 5);
  if (e >= nnz) return;
  int r = rows[e];
  int c = cols[e];
  float v = vals[e];
  float xv = Xp[(long)c * 32 + lane];
  atomicAdd(acc + (long)r * 32 + lane, v * xv);
}

// ---------------------------------------------------------------- y = scale * relu(y), in place
__global__ __launch_bounds__(256) void relu_scale(float* __restrict__ p, long n, float s) {
  long i = (long)blockIdx.x * blockDim.x + threadIdx.x;
  if (i < n) {
    float v = p[i];
    p[i] = v > 0.0f ? s * v : 0.0f;
  }
}

// ---------------------------------------------------------------- pool: segment_sum(|x1|) by batch id
__global__ __launch_bounds__(256) void pool_kernel(
    const float* __restrict__ x1, const int* __restrict__ batch,
    float* __restrict__ sums, float* __restrict__ counts, int n)
{
  const int lane = threadIdx.x & 31;
  long e = (long)blockIdx.x * 8 + (threadIdx.x >> 5);
  if (e >= n) return;
  int b = batch[e];
  atomicAdd(sums + (long)b * 32 + lane, fabsf(x1[e * 32 + lane]));
  if (lane == 0) atomicAdd(counts + b, 1.0f);
}

// ---------------------------------------------------------------- final MLP + softmax (128 graphs)
__global__ __launch_bounds__(128) void mlp_kernel(
    const float* __restrict__ sums, const float* __restrict__ counts,
    const float* __restrict__ w1, const float* __restrict__ b1,
    const float* __restrict__ w2, const float* __restrict__ b2,
    float* __restrict__ out)
{
  int g = threadIdx.x;
  if (g >= NGC) return;
  float cnt = fmaxf(counts[g], 1.0f);
  float pooled[32], h[32];
#pragma unroll
  for (int j = 0; j < 32; ++j) pooled[j] = sums[g * 32 + j] / cnt;
#pragma unroll
  for (int j = 0; j < 32; ++j) {
    float s = b1[j];
    for (int k = 0; k < 32; ++k) s += pooled[k] * w1[k * 32 + j];
    h[j] = fmaxf(s, 0.0f);
  }
  float logits[10];
  float mx = -1e30f;
#pragma unroll
  for (int j = 0; j < 10; ++j) {
    float s = b2[j];
    for (int k = 0; k < 32; ++k) s += h[k] * w2[k * 10 + j];
    logits[j] = s;
    mx = fmaxf(mx, s);
  }
  float den = 0.0f;
#pragma unroll
  for (int j = 0; j < 10; ++j) { logits[j] = __expf(logits[j] - mx); den += logits[j]; }
#pragma unroll
  for (int j = 0; j < 10; ++j) out[g * 10 + j] = logits[j] / den;
}

// ================================================================ launcher
extern "C" void kernel_launch(void* const* d_in, const int* in_sizes, int n_in,
                              void* d_out, int out_size, void* d_ws, size_t ws_size,
                              hipStream_t stream)
{
  const float* X0 = (const float*)d_in[0];
  const float* X1 = (const float*)d_in[1];
  const float* X2 = (const float*)d_in[2];

  // sparse order: 0=L0 1=L1 2=L2 3=B2D3 4=D2B1TD1inv 5=D1invB1 6=B2TD2inv
  const int*   sp_rows[7];
  const int*   sp_cols[7];
  const float* sp_vals[7];
  int          sp_nnz[7];
  for (int i = 0; i < 7; ++i) {
    sp_rows[i] = (const int*)  d_in[3 + 3 * i];
    sp_cols[i] = (const int*)  d_in[4 + 3 * i];
    sp_vals[i] = (const float*)d_in[5 + 3 * i];
    sp_nnz[i]  = in_sizes[3 + 3 * i];
  }
  const int*   batch1 = (const int*)  d_in[24];
  const float* W1     = (const float*)d_in[25];   // [7,64,32]
  const float* W234   = (const float*)d_in[26];   // [3,7,32,32]
  const float* mlp1_w = (const float*)d_in[27];
  const float* mlp1_b = (const float*)d_in[28];
  const float* mlp2_w = (const float*)d_in[29];
  const float* mlp2_b = (const float*)d_in[30];

  // workspace layout (floats)
  float* ws = (float*)d_ws;
  size_t o = 0;
  float* A0 = ws + o; o += (size_t)N0C * 32;
  float* A1 = ws + o; o += (size_t)N1C * 32;
  float* A2 = ws + o; o += (size_t)N2C * 32;
  float* B0 = ws + o; o += (size_t)N0C * 32;
  float* B1 = ws + o; o += (size_t)N1C * 32;
  float* B2 = ws + o; o += (size_t)N2C * 32;
  float* proj = ws + o; o += (size_t)N1C * 32;   // largest source set
  float* psum = ws + o; o += (size_t)NGC * 32;
  float* pcnt = ws + o; o += (size_t)NGC;

  // per-op tables: W index order n2n,n2e,e2e,e2n,e2t,t2e,t2t
  const int  op_src[7] = {0, 0, 1, 1, 1, 2, 2};
  const int  op_sp [7] = {0, 4, 1, 5, 6, 3, 2};
  const int  op_dst[7] = {0, 1, 1, 0, 2, 1, 2};
  const long nN[3]     = {N0C, N1C, N2C};
  const float scl[3]   = {0.5f, 1.0f / 3.0f, 0.5f};

  const float* cur[3] = {X0, X1, X2};

  for (int layer = 0; layer < 4; ++layer) {
    const int K = (layer == 0) ? 64 : 32;
    const float* Wl = (layer == 0) ? W1 : (W234 + (size_t)(layer - 1) * 7 * 32 * 32);
    float* acc[3];
    if ((layer & 1) == 0) { acc[0] = A0; acc[1] = A1; acc[2] = A2; }
    else                  { acc[0] = B0; acc[1] = B1; acc[2] = B2; }

    for (int t = 0; t < 3; ++t)
      zero_f32<<<cdiv_l(nN[t] * 32, 256), 256, 0, stream>>>(acc[t], nN[t] * 32);

    for (int op = 0; op < 7; ++op) {
      const float* xsrc = cur[op_src[op]];
      const long   Ns   = nN[op_src[op]];
      const float* Wop  = Wl + (size_t)op * K * 32;
      gemm_wmma_f16<<<cdiv_l(Ns, 128), 256, 0, stream>>>(xsrc, Wop, proj, (int)Ns, K);
      const int s = op_sp[op];
      spmm_atomic<<<cdiv_l(sp_nnz[s], 8), 256, 0, stream>>>(
          sp_rows[s], sp_cols[s], sp_vals[s], proj, acc[op_dst[op]], sp_nnz[s]);
    }

    for (int t = 0; t < 3; ++t)
      relu_scale<<<cdiv_l(nN[t] * 32, 256), 256, 0, stream>>>(acc[t], nN[t] * 32, scl[t]);

    cur[0] = acc[0]; cur[1] = acc[1]; cur[2] = acc[2];
  }

  // global mean pool of |x1| over batch ids, then MLP + softmax
  zero_f32<<<cdiv_l((long)NGC * 32, 256), 256, 0, stream>>>(psum, (long)NGC * 32);
  zero_f32<<<1, 256, 0, stream>>>(pcnt, (long)NGC);
  pool_kernel<<<cdiv_l((long)N1C, 8), 256, 0, stream>>>(cur[1], batch1, psum, pcnt, N1C);
  mlp_kernel<<<1, 128, 0, stream>>>(psum, pcnt, mlp1_w, mlp1_b, mlp2_w, mlp2_b, (float*)d_out);
}
